// PointTransformerBlock_75419625718655
// MI455X (gfx1250) — compile-verified, matrix operand order checked by reading, not run on hardware
//
#include <hip/hip_runtime.h>

// ---------------------------------------------------------------------------
// Point Transformer block for MI455X (gfx1250), bf16 WMMA throughout.
// Attention stages K/V tiles in LDS via CDNA5 async global->LDS copies
// (ASYNCcnt), double-buffered to overlap with WMMA + online softmax.
// ---------------------------------------------------------------------------

#define B_     4
#define N_     2048
#define DIM_   384
#define HEADS_ 8
#define HD_    48
#define HDP_   64            // head_dim padded to 64 so K-loop is 2x K=32 WMMA
#define HID_   1536
#define ROWS_  (B_ * N_)     // 8192
#define BIG_NEG_ (-1e30f)

typedef __bf16 bf16;
typedef __attribute__((ext_vector_type(16))) __bf16 v16bf;
typedef __attribute__((ext_vector_type(8)))  __bf16 v8bf;
typedef __attribute__((ext_vector_type(8)))  float  v8f;

// Per-lane 16-bit A/B fragment (16x32 / 32x16): elements 0..7 are 8 contiguous
// values at p, elements 8..15 are the 8 values at p+16 (K-halves per ISA
// layout; caller offsets p by +8 for lanes 16..31).
static __device__ __forceinline__ v16bf load_frag16(const bf16* p) {
  v8bf lo = *reinterpret_cast<const v8bf*>(p);
  v8bf hi = *reinterpret_cast<const v8bf*>(p + 16);
  return __builtin_shufflevector(lo, hi, 0, 1, 2, 3, 4, 5, 6, 7,
                                         8, 9, 10, 11, 12, 13, 14, 15);
}

static __device__ __forceinline__ v8f wmma_bf16(v16bf a, v16bf b, v8f c) {
  return __builtin_amdgcn_wmma_f32_16x16x32_bf16(
      /*neg_a=*/false, a, /*neg_b=*/false, b,
      /*c_mod=*/(short)0, c, /*reuse_a=*/false, /*reuse_b=*/false);
}

static __device__ __forceinline__ float gelu_exact(float v) {
  return 0.5f * v * (1.0f + erff(v * 0.70710678118654752f));
}

// CDNA5 async global->LDS copy: per-lane 16B, tracked by ASYNCcnt.
static __device__ __forceinline__ void async_copy_b128(const bf16* gsrc,
                                                       const bf16* lds_dst) {
  // LDS generic addresses truncate to the LDS byte offset in the low 32 bits.
  unsigned loff = (unsigned)(uintptr_t)lds_dst;
  asm volatile("global_load_async_to_lds_b128 %0, %1, off"
               :: "v"(loff), "v"(gsrc)
               : "memory");
}

static __device__ __forceinline__ void wait_async0() {
  asm volatile("s_wait_asynccnt 0x0" ::: "memory");
}

// ---------------------------------------------------------------------------
// Weight transpose fp32 -> bf16: Wt[n*K + k] = W[k*N + n]
// ---------------------------------------------------------------------------
__global__ void k_transpose_bf16(const float* __restrict__ W,
                                 bf16* __restrict__ Wt, int K, int N) {
  int n = blockIdx.x * 16 + threadIdx.x;
  int k = blockIdx.y * 16 + threadIdx.y;
  if (n < N && k < K) Wt[(size_t)n * K + k] = (bf16)W[(size_t)k * N + n];
}

__global__ void k_zero_bf16(bf16* __restrict__ p, size_t n) {
  size_t i = (size_t)blockIdx.x * blockDim.x + threadIdx.x;
  if (i < n) p[i] = (bf16)0.0f;
}

// ---------------------------------------------------------------------------
// LayerNorm (128 threads per row of 384); WITH_POS additionally runs the
// positional MLP: pe = gelu(pos@pw1 + pb1) @ pw2 + pb2   (48-wide)
// ---------------------------------------------------------------------------
template <int WITH_POS>
__global__ void k_ln(const float* __restrict__ xin, const float* __restrict__ g,
                     const float* __restrict__ be, bf16* __restrict__ hout,
                     const float* __restrict__ pos, const float* __restrict__ pw1,
                     const float* __restrict__ pb1, const float* __restrict__ pw2,
                     const float* __restrict__ pb2, float* __restrict__ pe) {
  int row = blockIdx.x;
  int t = threadIdx.x;                 // 0..127
  int lane = t & 31, wid = t >> 5;
  const float* xr = xin + (size_t)row * DIM_;
  float v0 = xr[t], v1 = xr[t + 128], v2 = xr[t + 256];
  __shared__ float red[8];

  float s = v0 + v1 + v2;
#pragma unroll
  for (int o = 16; o > 0; o >>= 1) s += __shfl_down(s, o);
  if (lane == 0) red[wid] = s;
  __syncthreads();
  float mean = (red[0] + red[1] + red[2] + red[3]) * (1.0f / DIM_);

  float d0 = v0 - mean, d1 = v1 - mean, d2 = v2 - mean;
  float q = d0 * d0 + d1 * d1 + d2 * d2;
#pragma unroll
  for (int o = 16; o > 0; o >>= 1) q += __shfl_down(q, o);
  if (lane == 0) red[wid + 4] = q;
  __syncthreads();
  float var = (red[4] + red[5] + red[6] + red[7]) * (1.0f / DIM_);
  float rstd = rsqrtf(var + 1e-5f);

  bf16* hr = hout + (size_t)row * DIM_;
  hr[t]       = (bf16)(d0 * rstd * g[t]       + be[t]);
  hr[t + 128] = (bf16)(d1 * rstd * g[t + 128] + be[t + 128]);
  hr[t + 256] = (bf16)(d2 * rstd * g[t + 256] + be[t + 256]);

  if (WITH_POS) {
    __shared__ float ph[HD_];
    if (t < HD_) {
      float p0 = pos[(size_t)row * 3 + 0];
      float p1 = pos[(size_t)row * 3 + 1];
      float p2 = pos[(size_t)row * 3 + 2];
      float a = p0 * pw1[t] + p1 * pw1[HD_ + t] + p2 * pw1[2 * HD_ + t] + pb1[t];
      ph[t] = gelu_exact(a);
    }
    __syncthreads();
    if (t < HD_) {
      float acc = pb2[t];
#pragma unroll 8
      for (int j = 0; j < HD_; ++j) acc += ph[j] * pw2[(size_t)j * HD_ + t];
      pe[(size_t)row * HD_ + t] = acc;
    }
  }
}

// ---------------------------------------------------------------------------
// Wave-tile WMMA GEMM: out[M x N] = A[M x K](bf16) @ Bt[N x K]^T(bf16) + bias,
// with per-mode epilogue. Block = (32,4): 4 waves, each owns a 16x64 C tile.
// ---------------------------------------------------------------------------
enum { MODE_Q = 0, MODE_K, MODE_V, MODE_WO, MODE_FC1, MODE_FC2 };

template <int MODE>
__global__ void k_gemm(const bf16* __restrict__ A, const bf16* __restrict__ Bt,
                       const float* __restrict__ bias, void* __restrict__ outp,
                       const float* __restrict__ aux, const int* __restrict__ mask,
                       int K) {
  int lane = threadIdx.x;
  int l16 = lane & 15;
  int hi8 = (lane >> 4) << 3;                    // 0 or 8 (K-half / row-half)
  int m0 = (blockIdx.x * 4 + threadIdx.y) * 16;  // 16 rows per wave
  int n0 = blockIdx.y * 64;                      // 64 cols per wave

  v8f c[4] = {};
  const bf16* pa = A + (size_t)(m0 + l16) * K + hi8;
  for (int k0 = 0; k0 < K; k0 += 32) {
    __builtin_prefetch(pa + k0 + 128, 0, 0);     // global_prefetch_b8 ahead
    v16bf a = load_frag16(pa + k0);
#pragma unroll
    for (int t2 = 0; t2 < 4; ++t2) {
      const bf16* pb = Bt + (size_t)(n0 + t2 * 16 + l16) * K + k0 + hi8;
      __builtin_prefetch(pb + 128, 0, 0);
      c[t2] = wmma_bf16(a, load_frag16(pb), c[t2]);
    }
  }

#pragma unroll
  for (int t2 = 0; t2 < 4; ++t2) {
    int n = n0 + t2 * 16 + l16;
    float bn = bias[n];
#pragma unroll
    for (int e = 0; e < 8; ++e) {
      int m = m0 + e + hi8;                      // C rows: e (+8 for hi half)
      float v = c[t2][e] + bn;
      if (MODE == MODE_Q || MODE == MODE_K || MODE == MODE_V) {
        int b = m >> 11, tok = m & (N_ - 1);
        int hd = n / HD_, d = n - hd * HD_;
        if (MODE == MODE_K) v += aux[(size_t)m * HD_ + d];   // pe broadcast
        if (MODE == MODE_V) {                                // V^T: (b,h,48,N)
          ((bf16*)outp)[((size_t)(b * HEADS_ + hd) * HD_ + d) * N_ + tok] = (bf16)v;
        } else {                                             // Q/K: (b,h,N,64)
          ((bf16*)outp)[((size_t)(b * HEADS_ + hd) * N_ + tok) * HDP_ + d] = (bf16)v;
        }
      } else if (MODE == MODE_WO) {
        // x1 = shortcut + where(mask, attn@Wo + bo, 0)
        float r = aux[(size_t)m * DIM_ + n] + (mask[m] ? v : 0.0f);
        ((float*)outp)[(size_t)m * DIM_ + n] = r;
      } else if (MODE == MODE_FC1) {
        ((bf16*)outp)[(size_t)m * HID_ + n] = (bf16)gelu_exact(v);
      } else {  // MODE_FC2: out = where(mask, x1 + mlp, 0)
        ((float*)outp)[(size_t)m * DIM_ + n] =
            mask[m] ? (aux[(size_t)m * DIM_ + n] + v) : 0.0f;
      }
    }
  }
}

// ---------------------------------------------------------------------------
// Flash attention per (b, h): block = 4 waves sharing a 64-query supertile.
// K/V tiles (32 keys) are staged in LDS with async global->LDS copies,
// double-buffered so the copy for block i+1 overlaps compute of block i.
// Per 32-key block: 4 WMMAs for S, shuffle online softmax, P re-layout via
// LDS, 3 WMMAs for P@V. Never materializes the NxN matrix.
// ---------------------------------------------------------------------------
__global__ void k_attn(const bf16* __restrict__ qp, const bf16* __restrict__ kp,
                       const bf16* __restrict__ vt, const int* __restrict__ mask,
                       bf16* __restrict__ aout) {
  __shared__ __attribute__((aligned(16))) bf16 ksh[2][32][HDP_];  // 2 x 4 KB
  __shared__ __attribute__((aligned(16))) bf16 vsh[2][HD_][32];   // 2 x 3 KB
  __shared__ __attribute__((aligned(16))) bf16 plds[4][16][32];   // 4 x 1 KB

  int lane = threadIdx.x;
  int l16 = lane & 15;
  int hi8 = (lane >> 4) << 3;
  int wv = threadIdx.y;                         // wave id in block (0..3)
  int tid = wv * 32 + lane;                     // 0..127
  int h = blockIdx.y, b = blockIdx.z;
  size_t bh = (size_t)b * HEADS_ + h;
  const bf16* Q  = qp + bh * (size_t)N_ * HDP_;
  const bf16* Kp = kp + bh * (size_t)N_ * HDP_;
  const bf16* Vt = vt + bh * (size_t)HD_ * N_;
  const int* mb = mask + b * N_;
  int q0 = blockIdx.x * 64 + wv * 16;

  // Stage one 32-key block of K (32x64) and V^T (48x32) into LDS buffer `buf`.
  auto issue_tiles = [&](int buf, int n0) {
    for (int c = tid; c < 256; c += 128) {      // K: 256 x 16B chunks
      const bf16* g = Kp + (size_t)(n0 + (c >> 3)) * HDP_ + ((c & 7) << 3);
      async_copy_b128(g, &ksh[buf][c >> 3][(c & 7) << 3]);
    }
    if (tid < 96) {
      for (int c = tid; c < 192; c += 96) {     // V: 192 x 16B chunks
        const bf16* g = Vt + (size_t)(c >> 2) * N_ + n0 + ((c & 3) << 3);
        async_copy_b128(g, &vsh[buf][c >> 2][(c & 3) << 3]);
      }
    }
  };

  // Loop-invariant Q fragments (head dim 0..31 and 32..63).
  const bf16* pq = Q + (size_t)(q0 + l16) * HDP_ + hi8;
  v16bf qa0 = load_frag16(pq);
  v16bf qa1 = load_frag16(pq + 32);

  bool qv[8];
#pragma unroll
  for (int e = 0; e < 8; ++e) qv[e] = (mb[q0 + e + hi8] != 0);

  float mrow[8], lrow[8];
#pragma unroll
  for (int e = 0; e < 8; ++e) { mrow[e] = BIG_NEG_; lrow[e] = 0.0f; }
  v8f o0 = {}, o1 = {}, o2 = {};

  const float scale = 0.14433756729740643f;  // 48^-0.5

  issue_tiles(0, 0);
  for (int n0 = 0; n0 < N_; n0 += 32) {
    int cur = (n0 >> 5) & 1;
    wait_async0();                 // this wave's async copies done
    __syncthreads();               // whole block's copies visible
    if (n0 + 32 < N_) issue_tiles(cur ^ 1, n0 + 32);   // prefetch next block

    // --- S = Q @ K^T for two 16-key sub-tiles (K fragments from LDS) ------
    v8f s0 = {}, s1 = {};
    const bf16* pk0 = &ksh[cur][l16][hi8];
    const bf16* pk1 = &ksh[cur][16 + l16][hi8];
    s0 = wmma_bf16(qa0, load_frag16(pk0), s0);
    s0 = wmma_bf16(qa1, load_frag16(pk0 + 32), s0);
    s1 = wmma_bf16(qa0, load_frag16(pk1), s1);
    s1 = wmma_bf16(qa1, load_frag16(pk1 + 32), s1);

    bool kv0 = (mb[n0 + l16] != 0);
    bool kv1 = (mb[n0 + 16 + l16] != 0);
#pragma unroll
    for (int e = 0; e < 8; ++e) {
      s0[e] = (kv0 && qv[e]) ? s0[e] * scale : BIG_NEG_;
      s1[e] = (kv1 && qv[e]) ? s1[e] * scale : BIG_NEG_;
    }

    // --- online softmax (row reductions within each 16-lane half) --------
#pragma unroll
    for (int e = 0; e < 8; ++e) {
      float t = fmaxf(s0[e], s1[e]);
      t = fmaxf(t, __shfl_xor(t, 1));
      t = fmaxf(t, __shfl_xor(t, 2));
      t = fmaxf(t, __shfl_xor(t, 4));
      t = fmaxf(t, __shfl_xor(t, 8));
      float mnew = fmaxf(mrow[e], t);
      float corr = __expf(mrow[e] - mnew);
      float p0 = __expf(s0[e] - mnew);
      float p1 = __expf(s1[e] - mnew);
      float ps = p0 + p1;
      ps += __shfl_xor(ps, 1);
      ps += __shfl_xor(ps, 2);
      ps += __shfl_xor(ps, 4);
      ps += __shfl_xor(ps, 8);
      lrow[e] = lrow[e] * corr + ps;
      mrow[e] = mnew;
      o0[e] *= corr; o1[e] *= corr; o2[e] *= corr;
      s0[e] = p0; s1[e] = p1;
    }

    // --- P: C-fragment -> A-fragment via LDS (DS ops in-order per wave) ---
#pragma unroll
    for (int e = 0; e < 8; ++e) {
      int mrw = e + hi8;
      plds[wv][mrw][l16]      = (bf16)s0[e];
      plds[wv][mrw][16 + l16] = (bf16)s1[e];
    }
    v16bf pfrag = load_frag16(&plds[wv][l16][hi8]);

    // --- O += P @ V  (V^T fragments from LDS, rows = head-dim) -----------
    o0 = wmma_bf16(pfrag, load_frag16(&vsh[cur][l16][hi8]), o0);
    o1 = wmma_bf16(pfrag, load_frag16(&vsh[cur][16 + l16][hi8]), o1);
    o2 = wmma_bf16(pfrag, load_frag16(&vsh[cur][32 + l16][hi8]), o2);

    __syncthreads();               // all waves done reading buf before reuse
  }

  // Normalize and write (b, token, h*48 + d) as bf16 for the Wo GEMM.
#pragma unroll
  for (int e = 0; e < 8; ++e) {
    float inv = 1.0f / lrow[e];
    int row = q0 + e + hi8;
    bf16* orow = aout + ((size_t)b * N_ + row) * DIM_ + h * HD_;
    orow[l16]      = (bf16)(o0[e] * inv);
    orow[16 + l16] = (bf16)(o1[e] * inv);
    orow[32 + l16] = (bf16)(o2[e] * inv);
  }
}

// ---------------------------------------------------------------------------
extern "C" void kernel_launch(void* const* d_in, const int* in_sizes, int n_in,
                              void* d_out, int out_size, void* d_ws, size_t ws_size,
                              hipStream_t stream) {
  (void)in_sizes; (void)n_in; (void)out_size; (void)ws_size;
  const float* x   = (const float*)d_in[0];
  const float* pos = (const float*)d_in[1];
  const int*   msk = (const int*)d_in[2];
  const float* Wq  = (const float*)d_in[3];
  const float* bq  = (const float*)d_in[4];
  const float* Wk  = (const float*)d_in[5];
  const float* bk  = (const float*)d_in[6];
  const float* Wv  = (const float*)d_in[7];
  const float* bv  = (const float*)d_in[8];
  const float* pw1 = (const float*)d_in[9];
  const float* pb1 = (const float*)d_in[10];
  const float* pw2 = (const float*)d_in[11];
  const float* pb2 = (const float*)d_in[12];
  const float* Wo  = (const float*)d_in[13];
  const float* bo  = (const float*)d_in[14];
  const float* mw1 = (const float*)d_in[15];
  const float* mb1 = (const float*)d_in[16];
  const float* mw2 = (const float*)d_in[17];
  const float* mb2 = (const float*)d_in[18];
  const float* g1  = (const float*)d_in[19];
  const float* be1 = (const float*)d_in[20];
  const float* g2  = (const float*)d_in[21];
  const float* be2 = (const float*)d_in[22];

  char* w = (char*)d_ws;
  auto carve = [&](size_t bytes) -> char* {
    char* p = w;
    w += (bytes + 255) & ~(size_t)255;
    return p;
  };
  bf16* WqT   = (bf16*)carve((size_t)DIM_ * DIM_ * 2);
  bf16* WkT   = (bf16*)carve((size_t)DIM_ * DIM_ * 2);
  bf16* WvT   = (bf16*)carve((size_t)DIM_ * DIM_ * 2);
  bf16* WoT   = (bf16*)carve((size_t)DIM_ * DIM_ * 2);
  bf16* mw1T  = (bf16*)carve((size_t)HID_ * DIM_ * 2);
  bf16* mw2T  = (bf16*)carve((size_t)DIM_ * HID_ * 2);
  bf16* h_bf  = (bf16*)carve((size_t)ROWS_ * DIM_ * 2);
  float* pe   = (float*)carve((size_t)ROWS_ * HD_ * 4);
  bf16* qpad  = (bf16*)carve((size_t)B_ * HEADS_ * N_ * HDP_ * 2);
  bf16* kpad  = (bf16*)carve((size_t)B_ * HEADS_ * N_ * HDP_ * 2);
  bf16* vtr   = (bf16*)carve((size_t)B_ * HEADS_ * HD_ * N_ * 2);
  bf16* attnb = (bf16*)carve((size_t)ROWS_ * DIM_ * 2);
  float* x1   = (float*)carve((size_t)ROWS_ * DIM_ * 4);
  bf16* h2    = (bf16*)carve((size_t)ROWS_ * DIM_ * 2);
  bf16* a1    = (bf16*)carve((size_t)ROWS_ * HID_ * 2);

  dim3 tb(16, 16);
  k_transpose_bf16<<<dim3(DIM_ / 16, DIM_ / 16), tb, 0, stream>>>(Wq, WqT, DIM_, DIM_);
  k_transpose_bf16<<<dim3(DIM_ / 16, DIM_ / 16), tb, 0, stream>>>(Wk, WkT, DIM_, DIM_);
  k_transpose_bf16<<<dim3(DIM_ / 16, DIM_ / 16), tb, 0, stream>>>(Wv, WvT, DIM_, DIM_);
  k_transpose_bf16<<<dim3(DIM_ / 16, DIM_ / 16), tb, 0, stream>>>(Wo, WoT, DIM_, DIM_);
  k_transpose_bf16<<<dim3(HID_ / 16, DIM_ / 16), tb, 0, stream>>>(mw1, mw1T, DIM_, HID_);
  k_transpose_bf16<<<dim3(DIM_ / 16, HID_ / 16), tb, 0, stream>>>(mw2, mw2T, HID_, DIM_);

  size_t qn = (size_t)B_ * HEADS_ * N_ * HDP_;
  k_zero_bf16<<<(unsigned)((qn + 255) / 256), 256, 0, stream>>>(qpad, qn);
  k_zero_bf16<<<(unsigned)((qn + 255) / 256), 256, 0, stream>>>(kpad, qn);

  // LN1 + positional MLP
  k_ln<1><<<ROWS_, 128, 0, stream>>>(x, g1, be1, h_bf, pos, pw1, pb1, pw2, pb2, pe);

  // QKV projections (K adds pe, V stores transposed)
  dim3 gblk(32, 4);
  k_gemm<MODE_Q><<<dim3(ROWS_ / 64, DIM_ / 64), gblk, 0, stream>>>(
      h_bf, WqT, bq, qpad, nullptr, nullptr, DIM_);
  k_gemm<MODE_K><<<dim3(ROWS_ / 64, DIM_ / 64), gblk, 0, stream>>>(
      h_bf, WkT, bk, kpad, pe, nullptr, DIM_);
  k_gemm<MODE_V><<<dim3(ROWS_ / 64, DIM_ / 64), gblk, 0, stream>>>(
      h_bf, WvT, bv, vtr, nullptr, nullptr, DIM_);

  // Flash attention (async-LDS staged K/V)
  k_attn<<<dim3(N_ / 64, HEADS_, B_), gblk, 0, stream>>>(qpad, kpad, vtr, msk, attnb);

  // Output projection + mask + residual -> x1 (fp32)
  k_gemm<MODE_WO><<<dim3(ROWS_ / 64, DIM_ / 64), gblk, 0, stream>>>(
      attnb, WoT, bo, x1, x, msk, DIM_);

  // LN2
  k_ln<0><<<ROWS_, 128, 0, stream>>>(x1, g2, be2, h2, nullptr, nullptr, nullptr,
                                     nullptr, nullptr, nullptr);

  // MLP
  k_gemm<MODE_FC1><<<dim3(ROWS_ / 64, HID_ / 64), gblk, 0, stream>>>(
      h2, mw1T, mb1, a1, nullptr, nullptr, DIM_);
  k_gemm<MODE_FC2><<<dim3(ROWS_ / 64, DIM_ / 64), gblk, 0, stream>>>(
      a1, mw2T, mb2, d_out, x1, msk, HID_);
}
